// Graphconv_68015102099731
// MI455X (gfx1250) — matrix operand order repaired and checked by reading, not compiled, not run on hardware
//
#include <hip/hip_runtime.h>
#include <hip/hip_bf16.h>

// ---------- types ----------
typedef __attribute__((ext_vector_type(16))) __bf16   v16bf;
typedef __attribute__((ext_vector_type(8)))  float    v8f;
typedef __attribute__((ext_vector_type(4)))  float    f4v;
typedef __attribute__((ext_vector_type(4)))  unsigned uint4v;
typedef __attribute__((ext_vector_type(2)))  unsigned uint2v;

#define F_DIM   128
#define N_NODES 25000
#define N_EDGES 400000

union FragBF {
    v16bf  v;
    uint4v q[2];
};

static __device__ __forceinline__ unsigned short f2bf(float f) {
    unsigned u = __float_as_uint(f);
    unsigned r = u + 0x7FFFu + ((u >> 16) & 1u);   // round-to-nearest-even
    return (unsigned short)(r >> 16);
}

// orderable-uint encoding so float max == uint max
static __device__ __forceinline__ unsigned encf(float f) {
    unsigned b = __float_as_uint(f);
    return (b & 0x80000000u) ? ~b : (b | 0x80000000u);
}
static __device__ __forceinline__ float decf(unsigned e) {
    unsigned b = (e & 0x80000000u) ? (e ^ 0x80000000u) : ~e;
    return __uint_as_float(b);
}
#define ENC_NEG_INF 0x007FFFFFu   // encf(-inf)

// ---------- weight pre-swizzle into WMMA B-fragment order ----------
__global__ void swizzle_weights(const float* __restrict__ W,
                                unsigned short* __restrict__ dstW, int KT) {
    int total = KT * 8 * 512;
    for (int idx = blockIdx.x * blockDim.x + threadIdx.x; idx < total;
         idx += gridDim.x * blockDim.x) {
        int tile = idx >> 9;
        int rem  = idx & 511;
        int lane = rem >> 4;
        int e    = rem & 15;
        int kk   = tile >> 3;
        int nt   = tile & 7;
        int k    = kk * 32 + ((lane & 16) ? 8 : 0) + (e & 7) + ((e & 8) ? 16 : 0);
        int n    = nt * 16 + (lane & 15);
        dstW[idx] = f2bf(W[k * F_DIM + n]);
    }
}

// ---------- init: zero hv(=d_out), denom; amax = -inf ----------
__global__ void init_buffers(float* __restrict__ out, float* __restrict__ denom,
                             unsigned* __restrict__ amax) {
    int idx = blockIdx.x * blockDim.x + threadIdx.x;
    if (idx < N_NODES * (F_DIM / 4)) ((f4v*)out)[idx] = (f4v){0.f, 0.f, 0.f, 0.f};
    if (idx < N_NODES) { denom[idx] = 0.f; amax[idx] = ENC_NEG_INF; }
}

// ---------- node eps MLP: eps = relu(nf@We1+be1)@We2+be2 (16-node tiles) ----------
__global__ __launch_bounds__(128) void node_eps(
    const float* __restrict__ nf,
    const unsigned short* __restrict__ We1s, const float* __restrict__ be1,
    const float* __restrict__ We2, const float* __restrict__ be2,
    float* __restrict__ eps) {
    __shared__ __align__(16) unsigned short xt[16][136];
    __shared__ float sAcc[16];

    const int tid = threadIdx.x;
    const int v0  = blockIdx.x * 16;
    if (tid < 16) sAcc[tid] = 0.f;

    for (int idx = tid; idx < 16 * 128; idx += 128) {
        int r = idx >> 7, k = idx & 127;
        int node = v0 + r;
        float val = (node < N_NODES) ? nf[(size_t)node * F_DIM + k] : 0.f;
        xt[r][k] = f2bf(val);
    }
    __syncthreads();

    const int wave = tid >> 5;
    const int lane = tid & 31;
    const int row  = lane & 15;
    const int hi   = (lane >> 4) & 1;
    const int nt0 = wave * 2, nt1 = wave * 2 + 1;

    v8f acc0 = {}, acc1 = {};
    for (int kk = 0; kk < 4; ++kk) {
        FragBF a, b0, b1;
        a.q[0] = *(const uint4v*)&xt[row][kk * 32 + hi * 8];
        a.q[1] = *(const uint4v*)&xt[row][kk * 32 + hi * 8 + 16];
        const uint4v* pb0 = (const uint4v*)(We1s + ((kk * 8 + nt0) * 512 + lane * 16));
        const uint4v* pb1 = (const uint4v*)(We1s + ((kk * 8 + nt1) * 512 + lane * 16));
        b0.q[0] = pb0[0]; b0.q[1] = pb0[1];
        b1.q[0] = pb1[0]; b1.q[1] = pb1[1];
        acc0 = __builtin_amdgcn_wmma_f32_16x16x32_bf16(false, a.v, false, b0.v,
                                                       (short)0, acc0, false, false);
        acc1 = __builtin_amdgcn_wmma_f32_16x16x32_bf16(false, a.v, false, b1.v,
                                                       (short)0, acc1, false, false);
    }

    float cr[8];
    {
        int f0 = nt0 * 16 + (lane & 15), f1 = nt1 * 16 + (lane & 15);
        float bb0 = be1[f0], bb1 = be1[f1];
        float w0 = We2[f0], w1 = We2[f1];
#pragma unroll
        for (int r = 0; r < 8; ++r) {
            float y0 = acc0[r] + bb0; y0 = y0 > 0.f ? y0 : 0.f;
            float y1 = acc1[r] + bb1; y1 = y1 > 0.f ? y1 : 0.f;
            cr[r] = y0 * w0 + y1 * w1;
        }
    }
#pragma unroll
    for (int r = 0; r < 8; ++r)
#pragma unroll
        for (int off = 1; off < 16; off <<= 1)
            cr[r] += __shfl_xor(cr[r], off, 32);
    if ((lane & 15) == 0)
#pragma unroll
        for (int r = 0; r < 8; ++r) atomicAdd(&sAcc[hi * 8 + r], cr[r]);
    __syncthreads();
    if (tid < 16 && v0 + tid < N_NODES) eps[v0 + tid] = sAcc[tid] + be2[0];
}

// ---------- edge score MLP + segment max: 64-edge M-tile, 4 waves x (64M x 32N) ----------
__global__ __launch_bounds__(128) void edge_scores(
    const float* __restrict__ nf, const float* __restrict__ gh,
    const int* __restrict__ src, const int* __restrict__ dst,
    const unsigned short* __restrict__ Wa1s, const float* __restrict__ ba1,
    const float* __restrict__ Wa2, const float* __restrict__ ba2,
    float* __restrict__ aOut, unsigned* __restrict__ amax) {
    __shared__ __align__(16) unsigned short xt[64][392];   // bf16 [h_src|h_dst|gh], padded stride
    __shared__ __align__(16) float ghs[64 * 128];          // fp32 gh tile, async-staged
    __shared__ int sSrc[64], sDst[64];
    __shared__ float sAcc[64];

    const int tid = threadIdx.x;
    const int e0  = blockIdx.x * 64;

    if (tid < 64) {
        sSrc[tid] = src[e0 + tid];
        sDst[tid] = dst[e0 + tid];
        sAcc[tid] = 0.f;
    }

    // --- async-stage the contiguous gh tile (32KB) into LDS while we gather h_src/h_dst ---
    // 2048 x 16B transfers, 128 threads -> 16 per thread
#pragma unroll 1
    for (int it = 0; it < 16; ++it) {
        int idx = tid + it * 128;                       // 16B chunk index
        unsigned ldsAddr = (unsigned)(uintptr_t)(&ghs[idx * 4]);
        const float* gp = gh + (size_t)e0 * F_DIM + idx * 4;
        asm volatile("global_load_async_to_lds_b128 %0, %1, off"
                     :: "v"(ldsAddr), "v"(gp) : "memory");
    }

    __syncthreads();   // sSrc/sDst visible

    // --- gather h_src / h_dst rows (L2-resident), convert to bf16 ---
    // 64 edges x 64 float4 (32 src + 32 dst) = 4096 vec4, 32 per thread
    for (int it = 0; it < 32; ++it) {
        int idx = tid + it * 128;
        int e   = idx >> 6;
        int sub = idx & 63;
        int node = (sub < 32) ? sSrc[e] : sDst[e];
        int kf   = (sub & 31) * 4;
        f4v v = *(const f4v*)(nf + (size_t)node * F_DIM + kf);
        int kx = (sub < 32 ? 0 : 128) + kf;
        uint2v p;
        p[0] = (unsigned)f2bf(v[0]) | ((unsigned)f2bf(v[1]) << 16);
        p[1] = (unsigned)f2bf(v[2]) | ((unsigned)f2bf(v[3]) << 16);
        *(uint2v*)&xt[e][kx] = p;
    }

    // --- wait for our own async transfers, then convert gh tile to bf16 ---
    asm volatile("s_wait_asynccnt 0" ::: "memory");
    for (int it = 0; it < 16; ++it) {
        int idx = tid + it * 128;                       // same chunks this thread loaded
        int e  = idx >> 5;
        int kf = (idx & 31) * 4;
        f4v v = *(const f4v*)(&ghs[e * 128 + kf]);
        uint2v p;
        p[0] = (unsigned)f2bf(v[0]) | ((unsigned)f2bf(v[1]) << 16);
        p[1] = (unsigned)f2bf(v[2]) | ((unsigned)f2bf(v[3]) << 16);
        *(uint2v*)&xt[e][256 + kf] = p;
    }
    __syncthreads();

    const int wave = tid >> 5;
    const int lane = tid & 31;
    const int row  = lane & 15;
    const int hi   = (lane >> 4) & 1;
    const int nt0 = wave * 2, nt1 = wave * 2 + 1;

    v8f acc[4][2];
#pragma unroll
    for (int m = 0; m < 4; ++m) { acc[m][0] = (v8f){}; acc[m][1] = (v8f){}; }

#pragma unroll 1
    for (int kk = 0; kk < 12; ++kk) {
        FragBF b0, b1;
        const uint4v* pb0 = (const uint4v*)(Wa1s + ((kk * 8 + nt0) * 512 + lane * 16));
        const uint4v* pb1 = (const uint4v*)(Wa1s + ((kk * 8 + nt1) * 512 + lane * 16));
        b0.q[0] = pb0[0]; b0.q[1] = pb0[1];
        b1.q[0] = pb1[0]; b1.q[1] = pb1[1];
#pragma unroll
        for (int m = 0; m < 4; ++m) {
            FragBF a;
            a.q[0] = *(const uint4v*)&xt[m * 16 + row][kk * 32 + hi * 8];
            a.q[1] = *(const uint4v*)&xt[m * 16 + row][kk * 32 + hi * 8 + 16];
            acc[m][0] = __builtin_amdgcn_wmma_f32_16x16x32_bf16(
                false, a.v, false, b0.v, (short)0, acc[m][0], false, false);
            acc[m][1] = __builtin_amdgcn_wmma_f32_16x16x32_bf16(
                false, a.v, false, b1.v, (short)0, acc[m][1], false, false);
        }
    }

    // epilogue: relu(+bias), dot with Wa2, reduce to per-edge scalar
    {
        int f0 = nt0 * 16 + (lane & 15), f1 = nt1 * 16 + (lane & 15);
        float bb0 = ba1[f0], bb1 = ba1[f1];
        float w0 = Wa2[f0], w1 = Wa2[f1];
#pragma unroll
        for (int m = 0; m < 4; ++m) {
            float cr[8];
#pragma unroll
            for (int r = 0; r < 8; ++r) {
                float y0 = acc[m][0][r] + bb0; y0 = y0 > 0.f ? y0 : 0.f;
                float y1 = acc[m][1][r] + bb1; y1 = y1 > 0.f ? y1 : 0.f;
                cr[r] = y0 * w0 + y1 * w1;
            }
#pragma unroll
            for (int r = 0; r < 8; ++r)
#pragma unroll
                for (int off = 1; off < 16; off <<= 1)
                    cr[r] += __shfl_xor(cr[r], off, 32);
            if ((lane & 15) == 0)
#pragma unroll
                for (int r = 0; r < 8; ++r)
                    atomicAdd(&sAcc[m * 16 + hi * 8 + r], cr[r]);
        }
    }
    __syncthreads();

    if (tid < 64) {
        float aval = sAcc[tid] + ba2[0];
        aOut[e0 + tid] = aval;
        atomicMax(&amax[sDst[tid]], encf(aval));
    }
}

// ---------- softmax numerator + denominator ----------
__global__ void edge_softmax_part(const int* __restrict__ dst,
                                  const unsigned* __restrict__ amax,
                                  float* __restrict__ aEx,
                                  float* __restrict__ denom) {
    int e = blockIdx.x * blockDim.x + threadIdx.x;
    if (e >= N_EDGES) return;
    int d = dst[e];
    float ex = __expf(aEx[e] - decf(amax[d]));
    aEx[e] = ex;
    atomicAdd(&denom[d], ex);
}

// ---------- messages: hv[dst] += alpha * h_src * gh (float4 per thread) ----------
__global__ void edge_messages(const float* __restrict__ nf, const float* __restrict__ gh,
                              const int* __restrict__ src, const int* __restrict__ dst,
                              const float* __restrict__ ex, const float* __restrict__ denom,
                              float* __restrict__ out) {
    int idx = blockIdx.x * blockDim.x + threadIdx.x;
    if (idx >= N_EDGES * (F_DIM / 4)) return;
    int e = idx >> 5;
    int f = (idx & 31) * 4;
    int d = dst[e];
    float alpha = ex[e] / denom[d];
    f4v h = *(const f4v*)(nf + (size_t)src[e] * F_DIM + f);
    f4v g = *(const f4v*)(gh + (size_t)e * F_DIM + f);
    float* o = out + (size_t)d * F_DIM + f;
    atomicAdd(o + 0, alpha * h[0] * g[0]);
    atomicAdd(o + 1, alpha * h[1] * g[1]);
    atomicAdd(o + 2, alpha * h[2] * g[2]);
    atomicAdd(o + 3, alpha * h[3] * g[3]);
}

// ---------- out = (1+eps)*hv + node_feats (in place over hv) ----------
__global__ void finalize(const float* __restrict__ nf, const float* __restrict__ eps,
                         float* __restrict__ out) {
    int idx = blockIdx.x * blockDim.x + threadIdx.x;
    if (idx >= N_NODES * (F_DIM / 4)) return;
    int v = idx >> 5;
    float s = 1.f + eps[v];
    f4v h = ((f4v*)out)[idx];
    f4v n = ((const f4v*)nf)[idx];
    ((f4v*)out)[idx] = (f4v){s * h[0] + n[0], s * h[1] + n[1],
                             s * h[2] + n[2], s * h[3] + n[3]};
}

extern "C" void kernel_launch(void* const* d_in, const int* in_sizes, int n_in,
                              void* d_out, int out_size, void* d_ws, size_t ws_size,
                              hipStream_t stream) {
    const float* nf  = (const float*)d_in[0];
    const float* gh  = (const float*)d_in[1];
    const int*   src = (const int*)d_in[2];
    const int*   dst = (const int*)d_in[3];
    const float* Wa1 = (const float*)d_in[4];
    const float* ba1 = (const float*)d_in[5];
    const float* Wa2 = (const float*)d_in[6];
    const float* ba2 = (const float*)d_in[7];
    const float* We1 = (const float*)d_in[8];
    const float* be1 = (const float*)d_in[9];
    const float* We2 = (const float*)d_in[10];
    const float* be2 = (const float*)d_in[11];
    float* out = (float*)d_out;

    // workspace layout (bytes): ~2.0 MB total
    char* ws = (char*)d_ws;
    unsigned short* Wa1s = (unsigned short*)(ws + 0);        //  98304 B (384x128 bf16, swizzled)
    unsigned short* We1s = (unsigned short*)(ws + 98304);    //  32768 B (128x128 bf16, swizzled)
    float*    aEx   = (float*)(ws + 131072);                 // 1.6 MB edge scores -> exp
    unsigned* amax  = (unsigned*)(ws + 1731072);             // 100 KB segment max (encoded)
    float*    denom = (float*)(ws + 1831072);                // 100 KB
    float*    eps   = (float*)(ws + 1931072);                // 100 KB

    swizzle_weights<<<192, 256, 0, stream>>>(Wa1, Wa1s, 12);
    swizzle_weights<<<64, 256, 0, stream>>>(We1, We1s, 4);
    init_buffers<<<(N_NODES * (F_DIM / 4) + 255) / 256, 256, 0, stream>>>(out, denom, amax);
    node_eps<<<(N_NODES + 15) / 16, 128, 0, stream>>>(nf, We1s, be1, We2, be2, eps);
    edge_scores<<<N_EDGES / 64, 128, 0, stream>>>(nf, gh, src, dst, Wa1s, ba1, Wa2, ba2,
                                                  aEx, amax);
    edge_softmax_part<<<(N_EDGES + 255) / 256, 256, 0, stream>>>(dst, amax, aEx, denom);
    edge_messages<<<(N_EDGES * (F_DIM / 4) + 255) / 256, 256, 0, stream>>>(
        nf, gh, src, dst, aEx, denom, out);
    finalize<<<(N_NODES * (F_DIM / 4) + 255) / 256, 256, 0, stream>>>(nf, eps, out);
}
